// resgated_multidigraph_88287347737111
// MI455X (gfx1250) — matrix operand order repaired and checked by reading, not compiled, not run on hardware
//
#include <hip/hip_runtime.h>

// ---------------------------------------------------------------------------
// GatedGCN forward for MI455X (gfx1250, wave32, WMMA f16 -> f32 accumulate)
//  - all GEMMs via v_wmma_f32_16x16x32_f16 (16 KFLOP/instr path)
//  - async global->LDS (ASYNCcnt) for raw tile copies / gathers
//  - LDS rows padded to 72/200 halves: 16B-aligned b128 frags, bank-conflict
//    free A-fragment reads
// ---------------------------------------------------------------------------

typedef _Float16 half_t;
typedef __attribute__((ext_vector_type(16))) _Float16 v16h;
typedef __attribute__((ext_vector_type(8)))  _Float16 v8h;
typedef __attribute__((ext_vector_type(8)))  float    v8f;

// ---------------- helpers ----------------

__device__ __forceinline__ float wave_sum32(float v) {
#pragma unroll
  for (int m = 16; m >= 1; m >>= 1) v += __shfl_xor(v, m, 32);
  return v;
}

__device__ __forceinline__ v8f v8f_zero() {
  v8f c;
#pragma unroll
  for (int i = 0; i < 8; ++i) c[i] = 0.f;
  return c;
}

__device__ __forceinline__ v8h v8h_zero() {
  v8h z;
#pragma unroll
  for (int i = 0; i < 8; ++i) z[i] = (half_t)0.f;
  return z;
}

// Async global->LDS 16B copy (gfx1250 GLOBAL_LOAD_ASYNC_TO_LDS_B128).
// LDS operand: wave-relative LDS byte offset = low 32 bits of the generic
// (flat-aperture) pointer to the __shared__ location.
__device__ __forceinline__ void async_copy_b128(const half_t* g, half_t* lds) {
  unsigned ldsoff = (unsigned)(size_t)lds;
  asm volatile("global_load_async_to_lds_b128 %0, %1, off"
               :: "v"(ldsoff), "v"(g) : "memory");
}
__device__ __forceinline__ void async_wait0() {
  asm volatile("s_wait_asynccnt 0x0" ::: "memory");
}

// A fragment: 16x32 f16 tile at LDS row base `rowbase`, row stride `ld`
// (halves), K offset kb. Lanes 0-15 row M=lane: elems 0..7 = K{kb+8*hi..},
// elems 8..15 = K{kb+16+8*hi..} (ISA 7.12.2 16-bit A layout).
__device__ __forceinline__ v16h load_a_frag(const half_t* rowbase, int ld, int kb) {
  int lane = threadIdx.x & 31;
  int m = lane & 15, hi = lane >> 4;
  const half_t* q = rowbase + m * ld + kb + 8 * hi;
  v8h c0 = *(const v8h*)q;
  v8h c1 = *(const v8h*)(q + 16);
  v16h a;
#pragma unroll
  for (int i = 0; i < 8; ++i) { a[i] = c0[i]; a[i + 8] = c1[i]; }
  return a;
}

// B fragment: 32x16 f16 of W (stored TRANSPOSED in LDS: wt[n][k], stride ld).
// Lane elem i<8 = W[kb + 8*hi + i][n], i>=8 = W[kb + 16 + 8*hi + (i-8)][n], n = nb*16 + (lane&15).
__device__ __forceinline__ v16h load_b_frag(const half_t* wt, int ld, int nb, int kb) {
  int lane = threadIdx.x & 31;
  int n = lane & 15, hi = lane >> 4;
  const half_t* q = wt + (nb * 16 + n) * ld + kb + 8 * hi;
  v8h c0 = *(const v8h*)q;
  v8h c1 = *(const v8h*)(q + 16);
  v16h b;
#pragma unroll
  for (int i = 0; i < 8; ++i) { b[i] = c0[i]; b[i + 8] = c1[i]; }
  return b;
}

__device__ __forceinline__ v8f wmma16x16x32(v16h a, v16h b, v8f c) {
  return __builtin_amdgcn_wmma_f32_16x16x32_f16(false, a, false, b,
                                                (short)0, c, false, false);
}

// ---------------- kernel 0: zero ----------------

__global__ void k_zero(float* __restrict__ p, int n) {
  int i = blockIdx.x * blockDim.x + threadIdx.x;
  if (i < n) p[i] = 0.f;
}

// ---------------- kernel 1: node input MLP ----------------
// h0 = W12( LN1( relu(W11(x)) ) ); one wave per node, lane owns feats f,f+32.

__global__ void __launch_bounds__(256) k_node_mlp(
    const float* __restrict__ x,
    const float* __restrict__ w11, const float* __restrict__ b11,
    const float* __restrict__ w12, const float* __restrict__ b12,
    const float* __restrict__ g1, const float* __restrict__ be1,
    float* __restrict__ h0f, half_t* __restrict__ h0h, int N) {
  int wv = threadIdx.x >> 5, lane = threadIdx.x & 31;
  int node = blockIdx.x * 8 + wv;
  if (node >= N) return;
  float x0 = x[node * 2 + 0], x1 = x[node * 2 + 1];
  int f0 = lane, f1 = lane + 32;
  float u0 = fmaxf(0.f, x0 * w11[f0] + x1 * w11[64 + f0] + b11[f0]);
  float u1 = fmaxf(0.f, x0 * w11[f1] + x1 * w11[64 + f1] + b11[f1]);
  float mean = wave_sum32(u0 + u1) * (1.f / 64.f);
  float d0 = u0 - mean, d1 = u1 - mean;
  float var = wave_sum32(d0 * d0 + d1 * d1) * (1.f / 64.f);
  float rstd = rsqrtf(var + 1e-5f);
  float ln0 = d0 * rstd * g1[f0] + be1[f0];
  float ln1 = d1 * rstd * g1[f1] + be1[f1];
  float acc0 = b12[f0], acc1 = b12[f1];
#pragma unroll
  for (int k = 0; k < 32; ++k) {
    float l = __shfl(ln0, k, 32);
    acc0 = fmaf(l, w12[k * 64 + f0], acc0);
    acc1 = fmaf(l, w12[k * 64 + f1], acc1);
  }
#pragma unroll
  for (int k = 0; k < 32; ++k) {
    float l = __shfl(ln1, k, 32);
    acc0 = fmaf(l, w12[(k + 32) * 64 + f0], acc0);
    acc1 = fmaf(l, w12[(k + 32) * 64 + f1], acc1);
  }
  h0f[node * 64 + f0] = acc0;
  h0f[node * 64 + f1] = acc1;
  h0h[node * 64 + f0] = (half_t)acc0;
  h0h[node * 64 + f1] = (half_t)acc1;
}

// ---------------- kernel 2: five fused node GEMMs (WMMA) ----------------
// A1h,A2h,A3h,B2h,B3h = h0 @ {A1,A2,A3,B2,B3} + b.  128 nodes / block,
// one A-fragment pair shared across all 5 weight matrices.

__global__ void __launch_bounds__(256) k_node_gemms(
    const half_t* __restrict__ h0h,
    const float* __restrict__ w0, const float* __restrict__ c0, float* __restrict__ o0,
    const float* __restrict__ w1, const float* __restrict__ c1, float* __restrict__ o1,
    const float* __restrict__ w2, const float* __restrict__ c2, float* __restrict__ o2,
    const float* __restrict__ w3, const float* __restrict__ c3, float* __restrict__ o3,
    const float* __restrict__ w4, const float* __restrict__ c4, float* __restrict__ o4,
    int N) {
  __shared__ __align__(16) half_t Xs[128][72];
  __shared__ __align__(16) half_t Wt[5][64][72];
  const float* ws[5] = {w0, w1, w2, w3, w4};
  const float* bs[5] = {c0, c1, c2, c3, c4};
  float* os[5] = {o0, o1, o2, o3, o4};
  int tid = threadIdx.x;
  int nbase = blockIdx.x * 128;
  bool full = (nbase + 128 <= N);
  int lane = tid & 31;
  int n_lo = lane & 15, hi = lane >> 4;

  // preload per-lane biases (hides latency behind staging)
  float biasr[5][4];
#pragma unroll
  for (int m = 0; m < 5; ++m)
#pragma unroll
    for (int nb = 0; nb < 4; ++nb) biasr[m][nb] = bs[m][nb * 16 + n_lo];

  // stage X tile: async global->LDS for full tiles
  if (full) {
    for (int idx = tid; idx < 128 * 8; idx += 256) {
      int r = idx >> 3, c = idx & 7;
      async_copy_b128(h0h + (size_t)(nbase + r) * 64 + c * 8, &Xs[r][c * 8]);
    }
  } else {
    for (int idx = tid; idx < 128 * 8; idx += 256) {
      int r = idx >> 3, c = idx & 7;
      int node = nbase + r;
      v8h v = (node < N) ? *(const v8h*)(h0h + (size_t)node * 64 + c * 8)
                         : v8h_zero();
      *(v8h*)&Xs[r][c * 8] = v;
    }
  }
  // stage 5 weight matrices, converted f32->f16, transposed
  for (int idx = tid; idx < 5 * 64 * 64; idx += 256) {
    int m = idx >> 12, r = idx & 4095;
    int k = r >> 6, n = r & 63;
    Wt[m][n][k] = (half_t)ws[m][r];
  }
  if (full) async_wait0();
  __syncthreads();

  int wv = tid >> 5;
  int row0 = wv * 16;
  v16h a0 = load_a_frag(&Xs[row0][0], 72, 0);
  v16h a1 = load_a_frag(&Xs[row0][0], 72, 32);
#pragma unroll
  for (int m = 0; m < 5; ++m) {
    v8f acc[4];
#pragma unroll
    for (int nb = 0; nb < 4; ++nb) acc[nb] = v8f_zero();
#pragma unroll
    for (int nb = 0; nb < 4; ++nb) {
      acc[nb] = wmma16x16x32(a0, load_b_frag(&Wt[m][0][0], 72, nb, 0), acc[nb]);
      acc[nb] = wmma16x16x32(a1, load_b_frag(&Wt[m][0][0], 72, nb, 32), acc[nb]);
    }
    if (full) {
      float* op = os[m] + (size_t)(nbase + row0 + 8 * hi) * 64 + n_lo;
#pragma unroll
      for (int nb = 0; nb < 4; ++nb)
#pragma unroll
        for (int j = 0; j < 8; ++j)
          op[(size_t)j * 64 + nb * 16] = acc[nb][j] + biasr[m][nb];
    } else {
#pragma unroll
      for (int nb = 0; nb < 4; ++nb)
#pragma unroll
        for (int j = 0; j < 8; ++j) {
          int node = nbase + row0 + j + 8 * hi;
          if (node < N)
            os[m][(size_t)node * 64 + nb * 16 + n_lo] = acc[nb][j] + biasr[m][nb];
        }
    }
  }
}

// ---------------- kernel 3: fused edge input (two chained WMMA GEMMs) ------
// ee = W22( LN2( relu(W21(e)) ) );  B1h = B1(ee).  128 edges / block.

__global__ void __launch_bounds__(256) k_edge_input(
    const float* __restrict__ e,
    const float* __restrict__ w21, const float* __restrict__ b21,
    const float* __restrict__ g2, const float* __restrict__ be2,
    const float* __restrict__ w22, const float* __restrict__ b22,
    const float* __restrict__ wB1, const float* __restrict__ bB1,
    half_t* __restrict__ eeh, half_t* __restrict__ b1hh, int E) {
  __shared__ __align__(16) half_t Xs[128][72];
  __shared__ __align__(16) half_t W22t[64][72];
  __shared__ __align__(16) half_t B1t[64][72];
  int tid = threadIdx.x;
  int ebase = blockIdx.x * 128;
  bool full = (ebase + 128 <= E);
  int wv = tid >> 5, lane = tid & 31;
  int n_lo = lane & 15, hi = lane >> 4;

  float bias22[4], biasB1[4];
#pragma unroll
  for (int nb = 0; nb < 4; ++nb) {
    bias22[nb] = b22[nb * 16 + n_lo];
    biasB1[nb] = bB1[nb * 16 + n_lo];
  }

  for (int idx = tid; idx < 4096; idx += 256) {
    int k = idx >> 6, n = idx & 63;
    W22t[n][k] = (half_t)w22[idx];
    B1t[n][k] = (half_t)wB1[idx];
  }
  int f0 = lane, f1 = lane + 32;
  float wa = w21[f0], wb = w21[f1], ba = b21[f0], bb = b21[f1];
  float ga = g2[f0], gb = g2[f1], ea = be2[f0], eb = be2[f1];
  for (int it = 0; it < 16; ++it) {
    int row = wv * 16 + it;
    int edge = ebase + row;
    float ev = (edge < E) ? e[edge] : 0.f;
    float u0 = fmaxf(0.f, ev * wa + ba);
    float u1 = fmaxf(0.f, ev * wb + bb);
    float mean = wave_sum32(u0 + u1) * (1.f / 64.f);
    float d0 = u0 - mean, d1 = u1 - mean;
    float var = wave_sum32(d0 * d0 + d1 * d1) * (1.f / 64.f);
    float rstd = rsqrtf(var + 1e-5f);
    Xs[row][f0] = (half_t)(d0 * rstd * ga + ea);
    Xs[row][f1] = (half_t)(d1 * rstd * gb + eb);
  }
  __syncthreads();
  int row0 = wv * 16;
  // GEMM 1: ee = LN-tile @ W22 + b22
  v16h a0 = load_a_frag(&Xs[row0][0], 72, 0);
  v16h a1 = load_a_frag(&Xs[row0][0], 72, 32);
  v8f acc[4];
#pragma unroll
  for (int nb = 0; nb < 4; ++nb) acc[nb] = v8f_zero();
#pragma unroll
  for (int nb = 0; nb < 4; ++nb) {
    acc[nb] = wmma16x16x32(a0, load_b_frag(&W22t[0][0], 72, nb, 0), acc[nb]);
    acc[nb] = wmma16x16x32(a1, load_b_frag(&W22t[0][0], 72, nb, 32), acc[nb]);
  }
  // store ee (f16) and stage it in-place for GEMM 2 (wave-local rows; LDS
  // ops from one wave stay ordered).
  {
    half_t* ep = eeh + (size_t)(ebase + row0 + 8 * hi) * 64 + n_lo;
#pragma unroll
    for (int nb = 0; nb < 4; ++nb)
#pragma unroll
      for (int j = 0; j < 8; ++j) {
        int row = row0 + j + 8 * hi;
        half_t hv = (half_t)(acc[nb][j] + bias22[nb]);
        if (full) ep[(size_t)j * 64 + nb * 16] = hv;
        else if (ebase + row < E) eeh[(size_t)(ebase + row) * 64 + nb * 16 + n_lo] = hv;
        Xs[row][nb * 16 + n_lo] = hv;
      }
  }
  // GEMM 2: B1h = ee @ B1 + bB1
  v16h a0b = load_a_frag(&Xs[row0][0], 72, 0);
  v16h a1b = load_a_frag(&Xs[row0][0], 72, 32);
  v8f ac2[4];
#pragma unroll
  for (int nb = 0; nb < 4; ++nb) ac2[nb] = v8f_zero();
#pragma unroll
  for (int nb = 0; nb < 4; ++nb) {
    ac2[nb] = wmma16x16x32(a0b, load_b_frag(&B1t[0][0], 72, nb, 0), ac2[nb]);
    ac2[nb] = wmma16x16x32(a1b, load_b_frag(&B1t[0][0], 72, nb, 32), ac2[nb]);
  }
  {
    half_t* bp = b1hh + (size_t)(ebase + row0 + 8 * hi) * 64 + n_lo;
#pragma unroll
    for (int nb = 0; nb < 4; ++nb)
#pragma unroll
      for (int j = 0; j < 8; ++j) {
        half_t hv = (half_t)(ac2[nb][j] + biasB1[nb]);
        if (full) bp[(size_t)j * 64 + nb * 16] = hv;
        else {
          int edge = ebase + row0 + j + 8 * hi;
          if (edge < E) b1hh[(size_t)edge * 64 + nb * 16 + n_lo] = hv;
        }
      }
  }
}

// ---------------- kernel 4: edge gate + message scatter ----------------
// One wave per edge; gathers hit L2 (node arrays are 12.8 MB << 192 MB L2).

__global__ void __launch_bounds__(256) k_edge_gate(
    const half_t* __restrict__ b1hh, const half_t* __restrict__ eeh,
    const float* __restrict__ A2h, const float* __restrict__ A3h,
    const float* __restrict__ B2h, const float* __restrict__ B3h,
    const int* __restrict__ src, const int* __restrict__ dst,
    const float* __restrict__ ge, const float* __restrict__ be,
    half_t* __restrict__ efwh, float* __restrict__ agg, int E) {
  int wv = threadIdx.x >> 5, lane = threadIdx.x & 31;
  int edge = blockIdx.x * 8 + wv;
  if (edge >= E) return;
  int s = src[edge], d = dst[edge];
  int f0 = lane, f1 = lane + 32;
  size_t eo = (size_t)edge * 64, so = (size_t)s * 64, doo = (size_t)d * 64;
  float b10 = (float)b1hh[eo + f0], b11v = (float)b1hh[eo + f1];
  float fw0 = fmaxf(0.f, b10 + B2h[so + f0] + B3h[doo + f0]);
  float fw1 = fmaxf(0.f, b11v + B2h[so + f1] + B3h[doo + f1]);
  float bw0 = fmaxf(0.f, b10 + B2h[doo + f0] + B3h[so + f0]);
  float bw1 = fmaxf(0.f, b11v + B2h[doo + f1] + B3h[so + f1]);
  float mf = wave_sum32(fw0 + fw1) * (1.f / 64.f);
  float df0 = fw0 - mf, df1 = fw1 - mf;
  float rf = rsqrtf(wave_sum32(df0 * df0 + df1 * df1) * (1.f / 64.f) + 1e-5f);
  float mb = wave_sum32(bw0 + bw1) * (1.f / 64.f);
  float db0 = bw0 - mb, db1 = bw1 - mb;
  float rb = rsqrtf(wave_sum32(db0 * db0 + db1 * db1) * (1.f / 64.f) + 1e-5f);
  float g0 = ge[f0], g1 = ge[f1], c0 = be[f0], c1 = be[f1];
  float ee0 = (float)eeh[eo + f0], ee1 = (float)eeh[eo + f1];
  float efw0 = ee0 + df0 * rf * g0 + c0;
  float efw1 = ee1 + df1 * rf * g1 + c1;
  float ebw0 = ee0 + db0 * rb * g0 + c0;
  float ebw1 = ee1 + db1 * rb * g1 + c1;
  efwh[eo + f0] = (half_t)efw0;
  efwh[eo + f1] = (half_t)efw1;
  float sf0 = 1.f / (1.f + __expf(-efw0));
  float sf1 = 1.f / (1.f + __expf(-efw1));
  float sb0 = 1.f / (1.f + __expf(-ebw0));
  float sb1 = 1.f / (1.f + __expf(-ebw1));
  float invf = 1.f / (wave_sum32(sf0 + sf1) + 1e-6f);
  float invb = 1.f / (wave_sum32(sb0 + sb1) + 1e-6f);
  float m0 = A2h[so + f0] * sf0 * invf + A3h[so + f0] * sb0 * invb;
  float m1 = A2h[so + f1] * sf1 * invf + A3h[so + f1] * sb1 * invb;
  atomicAdd(&agg[doo + f0], m0);
  atomicAdd(&agg[doo + f1], m1);
}

// ---------------- kernel 5: node update ----------------
// h = h0 + LN_h(relu(A1h + agg)); store as f16 for the scorer GEMM.

__global__ void __launch_bounds__(256) k_node_update(
    const float* __restrict__ A1h, const float* __restrict__ agg,
    const float* __restrict__ h0f,
    const float* __restrict__ gh, const float* __restrict__ bh,
    half_t* __restrict__ hh, int N) {
  int wv = threadIdx.x >> 5, lane = threadIdx.x & 31;
  int node = blockIdx.x * 8 + wv;
  if (node >= N) return;
  int f0 = lane, f1 = lane + 32;
  size_t no = (size_t)node * 64;
  float v0 = fmaxf(0.f, A1h[no + f0] + agg[no + f0]);
  float v1 = fmaxf(0.f, A1h[no + f1] + agg[no + f1]);
  float mean = wave_sum32(v0 + v1) * (1.f / 64.f);
  float d0 = v0 - mean, d1 = v1 - mean;
  float rstd = rsqrtf(wave_sum32(d0 * d0 + d1 * d1) * (1.f / 64.f) + 1e-5f);
  float h0 = h0f[no + f0] + d0 * rstd * gh[f0] + bh[f0];
  float h1 = h0f[no + f1] + d1 * rstd * gh[f1] + bh[f1];
  hh[no + f0] = (half_t)h0;
  hh[no + f1] = (half_t)h1;
}

// ---------------- kernel 6: edge scorer (K=192 WMMA + row dot) ----------
// Gathers concat(h[src], h[dst], e_fw) rows into LDS via async global->LDS.

__global__ void __launch_bounds__(128) k_scorer(
    const half_t* __restrict__ hh, const half_t* __restrict__ efwh,
    const int* __restrict__ src, const int* __restrict__ dst,
    const float* __restrict__ w1, const float* __restrict__ b1,
    const float* __restrict__ w2, const float* __restrict__ b2,
    float* __restrict__ out, int E) {
  __shared__ __align__(16) half_t S1t[64][200];  // scorer1 transposed [n][k]
  __shared__ __align__(16) half_t Xs[64][200];   // [edge][h_src|h_dst|e_fw]
  __shared__ float w2s[64];
  __shared__ int sids[64], dids[64];
  int tid = threadIdx.x;
  int ebase = blockIdx.x * 64;
  bool full = (ebase + 64 <= E);
  int wv = tid >> 5, lane = tid & 31;
  int n_lo = lane & 15, hi = lane >> 4;

  float bias1[4];
#pragma unroll
  for (int nb = 0; nb < 4; ++nb) bias1[nb] = b1[nb * 16 + n_lo];
  float b2v = b2[0];

  if (tid < 64) {
    int eg = ebase + tid;
    int cl = (eg < E) ? eg : 0;
    sids[tid] = src[cl];
    dids[tid] = dst[cl];
    w2s[tid] = w2[tid];
  }
  for (int idx = tid; idx < 192 * 64; idx += 128) {
    int k = idx >> 6, n = idx & 63;
    S1t[n][k] = (half_t)w1[idx];
  }
  __syncthreads();

  if (full) {
    for (int idx = tid; idx < 64 * 24; idx += 128) {
      int r = idx / 24, c = idx % 24;
      int seg = c >> 3, cc = c & 7;
      const half_t* sp =
          (seg == 0) ? hh + (size_t)sids[r] * 64 + cc * 8
        : (seg == 1) ? hh + (size_t)dids[r] * 64 + cc * 8
                     : efwh + (size_t)(ebase + r) * 64 + cc * 8;
      async_copy_b128(sp, &Xs[r][c * 8]);
    }
    async_wait0();
  } else {
    for (int idx = tid; idx < 64 * 24; idx += 128) {
      int r = idx / 24, c = idx % 24;
      int seg = c >> 3, cc = c & 7;
      v8h v;
      if (ebase + r < E) {
        const half_t* sp =
            (seg == 0) ? hh + (size_t)sids[r] * 64 + cc * 8
          : (seg == 1) ? hh + (size_t)dids[r] * 64 + cc * 8
                       : efwh + (size_t)(ebase + r) * 64 + cc * 8;
        v = *(const v8h*)sp;
      } else {
        v = v8h_zero();
      }
      *(v8h*)&Xs[r][c * 8] = v;
    }
  }
  __syncthreads();

  int row0 = wv * 16;
  v8f acc[4];
#pragma unroll
  for (int nb = 0; nb < 4; ++nb) acc[nb] = v8f_zero();
#pragma unroll
  for (int kb = 0; kb < 192; kb += 32) {
    v16h a = load_a_frag(&Xs[row0][0], 200, kb);
#pragma unroll
    for (int nb = 0; nb < 4; ++nb)
      acc[nb] = wmma16x16x32(a, load_b_frag(&S1t[0][0], 200, nb, kb), acc[nb]);
  }
  __syncthreads();  // all A-fragment reads done before we overlay hidden
  float* hid = (float*)&Xs[row0][0];  // wave-local rows reused as f32 scratch
#pragma unroll
  for (int nb = 0; nb < 4; ++nb) {
    int n = nb * 16 + n_lo;
#pragma unroll
    for (int j = 0; j < 8; ++j)
      hid[(j + 8 * hi) * 68 + n] = fmaxf(0.f, acc[nb][j] + bias1[nb]);
  }
  __syncthreads();
  int m = lane & 15, hseg = lane >> 4;
  float s = 0.f;
#pragma unroll
  for (int n = 0; n < 32; ++n) {
    int nn = hseg * 32 + n;
    s = fmaf(hid[m * 68 + nn], w2s[nn], s);
  }
  s += __shfl_xor(s, 16, 32);
  int edge = ebase + row0 + m;
  if (lane < 16 && edge < E) out[edge] = s + b2v;
}

// ---------------- host launch ----------------
// setup_inputs() order: x, e, src, dst, params{W11,W12,W21,W22,ln1,ln2,
//   A1,A2,A3,B1,B2,B3,ln_h,ln_e,scorer1,scorer2}; each linear = {w,b},
//   each ln = {g,b}  ->  36 flat inputs.

extern "C" void kernel_launch(void* const* d_in, const int* in_sizes, int n_in,
                              void* d_out, int out_size, void* d_ws, size_t ws_size,
                              hipStream_t stream) {
  const int N = in_sizes[0] / 2;  // NODE_F = 2
  const int E = in_sizes[1];      // EDGE_F = 1

  const float* x   = (const float*)d_in[0];
  const float* e   = (const float*)d_in[1];
  const int*   src = (const int*)d_in[2];
  const int*   dst = (const int*)d_in[3];
  const float* W11w = (const float*)d_in[4];  const float* W11b = (const float*)d_in[5];
  const float* W12w = (const float*)d_in[6];  const float* W12b = (const float*)d_in[7];
  const float* W21w = (const float*)d_in[8];  const float* W21b = (const float*)d_in[9];
  const float* W22w = (const float*)d_in[10]; const float* W22b = (const float*)d_in[11];
  const float* ln1g = (const float*)d_in[12]; const float* ln1b = (const float*)d_in[13];
  const float* ln2g = (const float*)d_in[14]; const float* ln2b = (const float*)d_in[15];
  const float* A1w = (const float*)d_in[16];  const float* A1b = (const float*)d_in[17];
  const float* A2w = (const float*)d_in[18];  const float* A2b = (const float*)d_in[19];
  const float* A3w = (const float*)d_in[20];  const float* A3b = (const float*)d_in[21];
  const float* B1w = (const float*)d_in[22];  const float* B1b = (const float*)d_in[23];
  const float* B2w = (const float*)d_in[24];  const float* B2b = (const float*)d_in[25];
  const float* B3w = (const float*)d_in[26];  const float* B3b = (const float*)d_in[27];
  const float* lnhg = (const float*)d_in[28]; const float* lnhb = (const float*)d_in[29];
  const float* lneg = (const float*)d_in[30]; const float* lneb = (const float*)d_in[31];
  const float* S1w = (const float*)d_in[32];  const float* S1b = (const float*)d_in[33];
  const float* S2w = (const float*)d_in[34];  const float* S2b = (const float*)d_in[35];
  float* out = (float*)d_out;

  // workspace carve-out (256B aligned blocks)
  char* base = (char*)d_ws;
  size_t off = 0;
  auto nxt = [&](size_t bytes) -> void* {
    off = (off + 255) & ~(size_t)255;
    void* r = base + off;
    off += bytes;
    return r;
  };
  float*  h0f  = (float*)nxt((size_t)N * 64 * 4);
  half_t* h0h  = (half_t*)nxt((size_t)N * 64 * 2);
  float*  A1h  = (float*)nxt((size_t)N * 64 * 4);
  float*  A2h  = (float*)nxt((size_t)N * 64 * 4);
  float*  A3h  = (float*)nxt((size_t)N * 64 * 4);
  float*  B2h  = (float*)nxt((size_t)N * 64 * 4);
  float*  B3h  = (float*)nxt((size_t)N * 64 * 4);
  float*  agg  = (float*)nxt((size_t)N * 64 * 4);
  half_t* hh   = (half_t*)nxt((size_t)N * 64 * 2);
  half_t* eeh  = (half_t*)nxt((size_t)E * 64 * 2);
  half_t* b1hh = (half_t*)nxt((size_t)E * 64 * 2);
  half_t* efwh = (half_t*)nxt((size_t)E * 64 * 2);
  (void)ws_size; (void)n_in; (void)out_size;

  {
    int tot = N * 64;
    k_zero<<<(tot + 255) / 256, 256, 0, stream>>>(agg, tot);
  }
  k_node_mlp<<<(N + 7) / 8, 256, 0, stream>>>(x, W11w, W11b, W12w, W12b,
                                              ln1g, ln1b, h0f, h0h, N);
  k_node_gemms<<<(N + 127) / 128, 256, 0, stream>>>(
      h0h, A1w, A1b, A1h, A2w, A2b, A2h, A3w, A3b, A3h,
      B2w, B2b, B2h, B3w, B3b, B3h, N);
  k_edge_input<<<(E + 127) / 128, 256, 0, stream>>>(
      e, W21w, W21b, ln2g, ln2b, W22w, W22b, B1w, B1b, eeh, b1hh, E);
  k_edge_gate<<<(E + 7) / 8, 256, 0, stream>>>(
      b1hh, eeh, A2h, A3h, B2h, B3h, src, dst, lneg, lneb, efwh, agg, E);
  k_node_update<<<(N + 7) / 8, 256, 0, stream>>>(A1h, agg, h0f, lnhg, lnhb, hh, N);
  k_scorer<<<(E + 63) / 64, 128, 0, stream>>>(
      hh, efwh, src, dst, S1w, S1b, S2w, S2b, out, E);
}
